// LSTMCellLayerNorm_12713103196909
// MI455X (gfx1250) — matrix operand order repaired and checked by reading, not compiled, not run on hardware
//
#include <hip/hip_runtime.h>
#include <hip/hip_bf16.h>

#define BATCH 4096
#define HID   1024
#define KTOT  2048
#define MT    64
#define NT    64
#define KSTEP 64
#define NSTAGE (KTOT / KSTEP)   // 32

typedef __attribute__((ext_vector_type(16))) __bf16 v16bf;
typedef __attribute__((ext_vector_type(8)))  float  v8f;

__device__ __forceinline__ unsigned short f32_bf16_rne(float f) {
    unsigned u = __float_as_uint(f);
    unsigned r = u + 0x7FFFu + ((u >> 16) & 1u);
    return (unsigned short)(r >> 16);
}

// gfx1250 async global->LDS copy, tracked by ASYNCcnt (no VGPR data staging).
__device__ __forceinline__ void async_copy_b128(unsigned lds_off, const unsigned short* g) {
    asm volatile("global_load_async_to_lds_b128 %0, %1, off"
                 :: "v"(lds_off), "v"(g) : "memory");
}
#define WAIT_ASYNC(n) asm volatile("s_wait_asynccnt " #n ::: "memory")

#if __has_builtin(__builtin_amdgcn_sched_group_barrier)
// masks: 0x100 = DS read, 0x008 = MFMA/WMMA
#define SCHED_GROUP(mask, cnt) __builtin_amdgcn_sched_group_barrier(mask, cnt, 0)
#else
#define SCHED_GROUP(mask, cnt)
#endif

// Kernel 1: pack [x|h] -> xh bf16 [4096 x 2048], [W_ih|W_hh] -> W bf16 [4096 x 2048]
__global__ __launch_bounds__(256)
void cvt_pack_kernel(const float* __restrict__ x, const float* __restrict__ h,
                     const float* __restrict__ Wih, const float* __restrict__ Whh,
                     unsigned short* __restrict__ xh, unsigned short* __restrict__ Wc) {
    const int QUADS = (BATCH * KTOT) / 4;  // per packed matrix
    int q = blockIdx.x * blockDim.x + threadIdx.x;
    const float* src0; const float* src1; unsigned short* dst;
    if (q < QUADS) { src0 = x; src1 = h; dst = xh; }
    else           { src0 = Wih; src1 = Whh; dst = Wc; q -= QUADS; }
    int e   = q << 2;
    int row = e >> 11;          // /2048
    int k   = e & (KTOT - 1);
    const float* s = (k < HID) ? (src0 + row * HID + k)
                               : (src1 + row * HID + (k - HID));
    float4 v = *(const float4*)s;
    unsigned short o[4] = { f32_bf16_rne(v.x), f32_bf16_rne(v.y),
                            f32_bf16_rne(v.z), f32_bf16_rne(v.w) };
    *(uint2*)(dst + (size_t)row * KTOT + k) = *(uint2*)o;
}

// Kernel 2: bf16 WMMA GEMM over all 4 gate blocks + fused gate math.
// Block = 256 threads (8 waves). Tile: 64 batch rows x 64 hidden cols x 4 gates.
// K=64 stages, double-buffered LDS filled by GLOBAL_LOAD_ASYNC_TO_LDS_B128.
__global__ __launch_bounds__(256)
void gemm_gates_kernel(const unsigned short* __restrict__ xh,   // [4096 x 2048] bf16
                       const unsigned short* __restrict__ Wc,   // [4096 x 2048] bf16
                       const float* __restrict__ c_in,
                       const float* __restrict__ b_ih, const float* __restrict__ b_hh,
                       float* __restrict__ c_pre, float* __restrict__ o_gate) {
    // Fragment-major LDS, double buffered, 2 K-halves per stage.
    __shared__ alignas(32) unsigned short lA[2][2][4][32][16];    // 16 KB
    __shared__ alignas(32) unsigned short lB[2][2][16][32][16];   // 64 KB
    const unsigned LA_BUF = 2u * 4u * 32u * 16u * 2u;             // 8192 B
    const unsigned LB_BUF = 2u * 16u * 32u * 16u * 2u;            // 32768 B
    const unsigned LA_KH  = 4u * 32u * 16u * 2u;                  // 4096 B
    const unsigned LB_KH  = 16u * 32u * 16u * 2u;                 // 16384 B

    const int tid  = threadIdx.x;
    const int lane = tid & 31;
    const int wave = tid >> 5;
    const int m0   = blockIdx.x * MT;
    const int n0   = blockIdx.y * NT;

    // ---- A staging mapping: one 16B chunk per thread per K-half ----
    // A frag (16-bit 16x32): lane<16 -> row=lane, k={0..7,16..23}; lane>=16 -> row=lane-16, k={8..15,24..31}
    const int a_row = tid >> 2;       // 0..63
    const int a_k8  = tid & 3;
    const unsigned short* a_src = xh + (size_t)(m0 + a_row) * KTOT + 8 * a_k8;
    const unsigned a_off =
        (unsigned)(size_t)&lA[0][0][a_row >> 4][(a_row & 15) + 16 * (a_k8 & 1)][8 * (a_k8 >> 1)];

    // ---- B staging mapping: four 16B chunks per thread per K-half ----
    // B frag (16-bit 32x16): lane<16 -> col n=lane, k={0..7,16..23}; lane>=16 -> col n=lane-16, k={8..15,24..31}
    const unsigned short* b_src[4];
    unsigned b_off[4];
    #pragma unroll
    for (int i = 0; i < 4; ++i) {
        int cid  = tid + 256 * i;     // 0..1023
        int sb   = cid >> 6;          // gate*4 + n_sub
        int c2   = cid & 63;
        int bl   = c2 >> 1;
        int half = c2 & 1;
        int gate = sb >> 2;
        int nsub = sb & 3;
        int nrow = gate * HID + n0 + 16 * nsub + (bl & 15);
        int kk   = 8 * (bl >> 4) + 16 * half;
        b_src[i] = Wc + (size_t)nrow * KTOT + kk;
        b_off[i] = (unsigned)(size_t)&lB[0][0][sb][bl][8 * half];
    }

    auto issue = [&](int stage, int buf) {
        const int kb = KSTEP * stage;
        #pragma unroll
        for (int kh = 0; kh < 2; ++kh)
            async_copy_b128(a_off + (unsigned)buf * LA_BUF + (unsigned)kh * LA_KH,
                            a_src + kb + 32 * kh);
        #pragma unroll
        for (int i = 0; i < 4; ++i)
            #pragma unroll
            for (int kh = 0; kh < 2; ++kh)
                async_copy_b128(b_off[i] + (unsigned)buf * LB_BUF + (unsigned)kh * LB_KH,
                                b_src[i] + kb + 32 * kh);
    };

    const int m_sub  = wave & 3;
    const int n_half = wave >> 2;

    v8f acc[8];   // acc[g*2+j]
    #pragma unroll
    for (int t = 0; t < 8; ++t)
        acc[t] = (v8f){0.f,0.f,0.f,0.f,0.f,0.f,0.f,0.f};

    issue(0, 0);
    issue(1, 1);

    for (int i = 0; i < NSTAGE; ++i) {
        const int p = i & 1;
        WAIT_ASYNC(10);          // stage i's 10 async ops landed (stage i+1's may remain)
        __syncthreads();         // all waves' buffer p is ready
        #pragma unroll
        for (int kh = 0; kh < 2; ++kh) {
            v16bf afrag = *(const v16bf*)&lA[p][kh][m_sub][lane][0];
            v16bf bf[8];
            #pragma unroll
            for (int t = 0; t < 8; ++t) {
                int sb = (t >> 1) * 4 + 2 * n_half + (t & 1);
                bf[t] = *(const v16bf*)&lB[p][kh][sb][lane][0];
            }
            #pragma unroll
            for (int t = 0; t < 8; ++t)
                acc[t] = __builtin_amdgcn_wmma_f32_16x16x32_bf16(
                    false, afrag, false, bf[t], (short)0, acc[t], false, false);
            // Pipeline this K-half: cluster the 18 fragment ds_loads (1 afrag = 2,
            // 8 bfrags = 16) ahead of the 8 back-to-back WMMAs.
            SCHED_GROUP(0x100, 18);
            SCHED_GROUP(0x008, 8);
        }
        __syncthreads();         // everyone done reading buffer p
        if (i + 2 < NSTAGE) issue(i + 2, p);
    }

    // ---- fused gate epilogue ----
    // C/D layout: VGPR r, lane l: m = r + (l>=16 ? 8 : 0), n = l & 15.
    float bias[4][2];
    int ncol[2];
    #pragma unroll
    for (int j = 0; j < 2; ++j)
        ncol[j] = n0 + 16 * (2 * n_half + j) + (lane & 15);
    #pragma unroll
    for (int g = 0; g < 4; ++g)
        #pragma unroll
        for (int j = 0; j < 2; ++j) {
            int col = g * HID + ncol[j];
            bias[g][j] = b_ih[col] + b_hh[col];
        }

    const int mbase = m0 + 16 * m_sub + ((lane >> 4) << 3);
    #pragma unroll
    for (int r = 0; r < 8; ++r) {
        int m = mbase + r;
        #pragma unroll
        for (int j = 0; j < 2; ++j) {
            int n = ncol[j];
            float pi = acc[0 * 2 + j][r] + bias[0][j];
            float pf = acc[1 * 2 + j][r] + bias[1][j];
            float po = acc[2 * 2 + j][r] + bias[2][j];
            float pg = acc[3 * 2 + j][r] + bias[3][j];
            float it = 1.f / (1.f + __expf(-pi));
            float ft = 1.f / (1.f + __expf(-pf));
            float ot = 1.f / (1.f + __expf(-po));
            float gt = tanhf(pg);
            size_t idx = (size_t)m * HID + n;
            c_pre[idx]  = c_in[idx] * ft + it * gt;
            o_gate[idx] = ot;
        }
    }
}

// Kernel 3: per-row LayerNorm + outputs. One 256-thread block per batch row.
__global__ __launch_bounds__(256)
void ln_out_kernel(const float* __restrict__ c_pre, const float* __restrict__ o_gate,
                   const float* __restrict__ gamma, const float* __restrict__ beta,
                   float* __restrict__ h_out, float* __restrict__ c_out) {
    const int row = blockIdx.x;
    const int tid = threadIdx.x;
    const size_t base = (size_t)row * HID;
    float4 v = *(const float4*)(c_pre + base + tid * 4);
    float s  = v.x + v.y + v.z + v.w;
    float s2 = v.x*v.x + v.y*v.y + v.z*v.z + v.w*v.w;
    #pragma unroll
    for (int off = 16; off > 0; off >>= 1) {
        s  += __shfl_xor(s,  off, 32);
        s2 += __shfl_xor(s2, off, 32);
    }
    __shared__ float ws1[8], ws2[8];
    __shared__ float mu_s, rstd_s;
    if ((tid & 31) == 0) { ws1[tid >> 5] = s; ws2[tid >> 5] = s2; }
    __syncthreads();
    if (tid == 0) {
        float t1 = 0.f, t2 = 0.f;
        #pragma unroll
        for (int i = 0; i < 8; ++i) { t1 += ws1[i]; t2 += ws2[i]; }
        float mu  = t1 * (1.0f / HID);
        float var = t2 * (1.0f / HID) - mu * mu;
        mu_s = mu; rstd_s = rsqrtf(var + 1e-5f);
    }
    __syncthreads();
    float mu = mu_s, rstd = rstd_s;
    float4 gm = *(const float4*)(gamma + tid * 4);
    float4 bt = *(const float4*)(beta  + tid * 4);
    float4 og = *(const float4*)(o_gate + base + tid * 4);
    float4 ct, ht;
    ct.x = (v.x - mu) * rstd * gm.x + bt.x;  ht.x = og.x * tanhf(ct.x);
    ct.y = (v.y - mu) * rstd * gm.y + bt.y;  ht.y = og.y * tanhf(ct.y);
    ct.z = (v.z - mu) * rstd * gm.z + bt.z;  ht.z = og.z * tanhf(ct.z);
    ct.w = (v.w - mu) * rstd * gm.w + bt.w;  ht.w = og.w * tanhf(ct.w);
    *(float4*)(c_out + base + tid * 4) = ct;
    *(float4*)(h_out + base + tid * 4) = ht;
}

extern "C" void kernel_launch(void* const* d_in, const int* in_sizes, int n_in,
                              void* d_out, int out_size, void* d_ws, size_t ws_size,
                              hipStream_t stream) {
    const float* x     = (const float*)d_in[0];
    const float* h     = (const float*)d_in[1];
    const float* c     = (const float*)d_in[2];
    const float* Wih   = (const float*)d_in[3];
    const float* bih   = (const float*)d_in[4];
    const float* Whh   = (const float*)d_in[5];
    const float* bhh   = (const float*)d_in[6];
    const float* gamma = (const float*)d_in[7];
    const float* beta  = (const float*)d_in[8];

    float* h_out = (float*)d_out;
    float* c_out = h_out + (size_t)BATCH * HID;

    char* ws = (char*)d_ws;
    unsigned short* xh = (unsigned short*)ws;                                  // 16 MB
    unsigned short* Wc = (unsigned short*)(ws + (size_t)BATCH * KTOT * 2);     // 16 MB
    float* c_pre  = (float*)(ws + (size_t)BATCH * KTOT * 4);                   // 16 MB
    float* o_gate = c_pre + (size_t)BATCH * HID;                               // 16 MB

    const int quads = (BATCH * KTOT) / 4;
    cvt_pack_kernel<<<(2 * quads) / 256, 256, 0, stream>>>(x, h, Wih, Whh, xh, Wc);

    dim3 g2(BATCH / MT, HID / NT);
    gemm_gates_kernel<<<g2, 256, 0, stream>>>(xh, Wc, c, bih, bhh, c_pre, o_gate);

    ln_out_kernel<<<BATCH, 256, 0, stream>>>(c_pre, o_gate, gamma, beta, h_out, c_out);
}
